// ScatteringNetwork_16862041604704
// MI455X (gfx1250) — matrix operand order repaired and checked by reading, not compiled, not run on hardware
//
#include <hip/hip_runtime.h>

typedef __attribute__((ext_vector_type(16))) _Float16 v16h;
typedef __attribute__((ext_vector_type(8)))  _Float16 v8h;
typedef __attribute__((ext_vector_type(8)))  float    v8f;

// ---------------- problem constants ----------------
#define NB   8          // batch
#define HW   256        // image H=W
#define NF   10         // wavelet filters
#define PSIK 11         // psi kernel size
#define BK   7          // blur kernel size
#define CK   17         // composed (psi (*) blur) kernel size
#define OUTW 32         // strided output H=W
#define NCH  111        // 1 + 10 + 100 output channels

#define S1_STEPS  6     // K = 11 rows x 16 slots = 176 -> 6 steps of 32
#define C17_STEPS 17    // K = 17 rows x 32 slots = 544 -> 17 steps of 32

// ---------------- workspace layout (bytes) ----------------
#define IMG_H_OFF   0u                 // f16 [8,256,256]        1,048,576 B
#define S1ABS_OFF   1048576u           // f16 [8,10,256,256]    10,485,760 B
#define BF1RE_OFF   11534336u          // f16 B-frag [6][32][16]     6,144 B
#define BF1IM_OFF   11540480u          // f16 B-frag [6][32][16]     6,144 B
#define BF2_OFF     11546624u          // f16 B-frag [17][32][16]   17,408 B
#define C1_OFF      11564032u          // f32 [10][17][17]          11,560 B

// B-matrix (32x16 f16) element location: lane L holds column n=L&15,
// elements 0-7 cover K=8*(L/16)+0..7, elements 8-15 cover K=16+8*(L/16)+0..7.
__device__ __forceinline__ int b_klocal(int lane, int h) {
    return ((h & 8) ? 16 : 0) + ((lane < 16) ? 0 : 8) + (h & 7);
}

// ---------------- prep: composed kernels + WMMA B fragments ----------------
__global__ __launch_bounds__(256)
void prep_kernel(const float* __restrict__ psi_re, const float* __restrict__ psi_im,
                 const float* __restrict__ blur,
                 float* __restrict__ C1,          // [10][17][17]
                 _Float16* __restrict__ bf1re, _Float16* __restrict__ bf1im,
                 _Float16* __restrict__ bf2)
{
    int tid = threadIdx.x;
    // Phase 1: composed kernel C1[j,u,v] = sum_{p,q} blur[p,q] * psi_re[j,u-p,v-q]
    for (int i = tid; i < NF * CK * CK; i += 256) {
        int j = i / (CK * CK), rem = i % (CK * CK);
        int u = rem / CK, v = rem % CK;
        float acc = 0.f;
        for (int p = 0; p < BK; ++p)
            for (int q = 0; q < BK; ++q) {
                int a = u - p, b = v - q;
                if (a >= 0 && a < PSIK && b >= 0 && b < PSIK)
                    acc += blur[p * BK + q] * psi_re[j * PSIK * PSIK + a * PSIK + b];
            }
        C1[i] = acc;
    }
    __syncthreads();
    // Phase 2: s1 B fragments. K layout: k = dy*16 + dx (dx slots 11..15 and
    // row 11 carry zero weights so A can be loaded as contiguous rows).
    for (int i = tid; i < S1_STEPS * 32 * 16; i += 256) {
        int s = i >> 9, lane = (i >> 4) & 31, h = i & 15;
        int n = lane & 15;
        int k = s * 32 + b_klocal(lane, h);
        int dy = k >> 4, dx = k & 15;
        float re = 0.f, im = 0.f;
        if (n < NF && dy < PSIK && dx < PSIK) {
            re = psi_re[n * PSIK * PSIK + dy * PSIK + dx];
            im = psi_im[n * PSIK * PSIK + dy * PSIK + dx];
        }
        bf1re[i] = (_Float16)re;
        bf1im[i] = (_Float16)im;
    }
    // Phase 3: composed-kernel B fragments. K layout: k = u*32 + v
    // (one 17-tap row per 32-K step; v slots 17..31 are zero weights).
    for (int i = tid; i < C17_STEPS * 32 * 16; i += 256) {
        int s = i >> 9, lane = (i >> 4) & 31, h = i & 15;
        int n = lane & 15;
        int k = s * 32 + b_klocal(lane, h);
        int v = k & 31;                       // u == s by construction
        float val = 0.f;
        if (n < NF && v < CK) val = C1[n * CK * CK + s * CK + v];
        bf2[i] = (_Float16)val;
    }
}

// ---------------- img f32 -> f16 ----------------
__global__ __launch_bounds__(256)
void img2h_kernel(const float* __restrict__ img, _Float16* __restrict__ img_h, int n)
{
    int i = blockIdx.x * blockDim.x + threadIdx.x;
    if (i < n) img_h[i] = (_Float16)img[i];
}

// ---------------- s0: 7x7 blur at stride 8 (tiny, VALU) ----------------
__global__ __launch_bounds__(256)
void s0_kernel(const float* __restrict__ img, const float* __restrict__ blur,
               float* __restrict__ out)
{
    int idx = blockIdx.x * blockDim.x + threadIdx.x;
    if (idx >= NB * OUTW * OUTW) return;
    int n = idx >> 10, p = idx & 1023;
    int yo = p >> 5, xo = p & 31;
    const float* plane = img + (size_t)n * HW * HW;
    float acc = 0.f;
    for (int u = 0; u < BK; ++u) {
        int iy = yo * 8 + u - 3;
        if (iy < 0 || iy >= HW) continue;
        for (int v = 0; v < BK; ++v) {
            int ix = xo * 8 + v - 3;
            if (ix < 0 || ix >= HW) continue;
            acc += plane[iy * HW + ix] * blur[u * BK + v];
        }
    }
    out[(size_t)n * NCH * 1024 + p] = acc;
}

// ---------------- s1: |img (*) psi| at full res, WMMA f16->f32 ----------------
// A fragment = two contiguous 8-half LDS reads per lane (rows r+2s, r+2s+1).
#define S1_LDSH 43            // rows r..r+11, r<=31
#define S1_LDSW 48            // 47 cols used (halo 5 left, 10+pad right)
__global__ __launch_bounds__(256)
void s1_wmma_kernel(const _Float16* __restrict__ img_h,
                    const _Float16* __restrict__ bf1re,
                    const _Float16* __restrict__ bf1im,
                    _Float16* __restrict__ s1_abs)
{
    __shared__ __align__(16) _Float16 tile[S1_LDSH * S1_LDSW];
    __shared__ __align__(16) _Float16 bre_l[S1_STEPS * 512];
    __shared__ __align__(16) _Float16 bim_l[S1_STEPS * 512];

    int n   = blockIdx.z;
    int ty0 = blockIdx.y * 32;
    int tx0 = blockIdx.x * 32;
    const _Float16* plane = img_h + ((size_t)n << 16);

    // stage 43x47 tile (rows ty0-5 .. ty0+37) with zero padding
    for (int i = threadIdx.x; i < S1_LDSH * 47; i += 256) {
        int r = i / 47, c = i % 47;
        int iy = ty0 + r - 5, ix = tx0 + c - 5;
        _Float16 v = (_Float16)0.f;
        if (iy >= 0 && iy < HW && ix >= 0 && ix < HW) v = plane[iy * HW + ix];
        tile[r * S1_LDSW + c] = v;
    }
    // stage B fragments (pre-packed) into LDS
    for (int i = threadIdx.x; i < S1_STEPS * 256; i += 256) {
        ((uint32_t*)bre_l)[i] = ((const uint32_t*)bf1re)[i];
        ((uint32_t*)bim_l)[i] = ((const uint32_t*)bf1im)[i];
    }
    __syncthreads();

    int lane = threadIdx.x & 31, wave = threadIdx.x >> 5;
    int mrow = lane & 15, off = (lane >> 4) * 8;

    for (int seg = wave; seg < 64; seg += 8) {
        int r    = seg >> 1;           // tile row
        int xpix = (seg & 1) * 16 + mrow;
        v8f cre = {}; v8f cim = {};
        #pragma unroll
        for (int s = 0; s < S1_STEPS; ++s) {
            const _Float16* p0 = &tile[(r + 2 * s) * S1_LDSW + xpix + off];
            v8h lo = *(const v8h*)p0;                  // K = s*32 + off + 0..7
            v8h hi = *(const v8h*)(p0 + S1_LDSW);      // K = s*32 + 16 + off + 0..7
            v16h a = __builtin_shufflevector(lo, hi, 0, 1, 2, 3, 4, 5, 6, 7,
                                             8, 9, 10, 11, 12, 13, 14, 15);
            v16h br = *(const v16h*)&bre_l[(s * 32 + lane) * 16];
            v16h bi = *(const v16h*)&bim_l[(s * 32 + lane) * 16];
            cre = __builtin_amdgcn_wmma_f32_16x16x32_f16(false, a, false, br,
                                                         (short)0, cre, false, false);
            cim = __builtin_amdgcn_wmma_f32_16x16x32_f16(false, a, false, bi,
                                                         (short)0, cim, false, false);
        }
        int ch = lane & 15;
        if (ch < NF) {
            _Float16* obase = s1_abs + (((size_t)(n * NF + ch)) << 16);
            int gy = ty0 + r;
            #pragma unroll
            for (int rr = 0; rr < 8; ++rr) {
                int m = rr + off;                       // pixel within segment
                float re = cre[rr], im = cim[rr];
                float ab = sqrtf(re * re + im * im);
                int gx = tx0 + (seg & 1) * 16 - mrow + xpix + (m - off) + 0;
                (void)gx;
                obase[gy * HW + tx0 + (seg & 1) * 16 + m] = (_Float16)ab;
            }
        }
    }
}

// ------- fused conv(17x17 composed kernel) + stride-8 sample, WMMA -------
// Output stride 8 makes every A-fragment LDS read genuinely 16B aligned.
// Used for s1_out (in=img_h, ncin=1, ch_base=1) and s2_out (in=|s1|, ncin=10, ch_base=11)
#define C17_ROWS 73
#define C17_COLS 288          // 280 cols used; 576B row stride (16B aligned)
__global__ __launch_bounds__(256)
void conv17s8_wmma_kernel(const _Float16* __restrict__ in_planes,
                          const _Float16* __restrict__ bf2,
                          float* __restrict__ out, int ncin, int ch_base)
{
    __shared__ __align__(16) _Float16 tile[C17_ROWS * C17_COLS];  // 42,048 B
    __shared__ __align__(16) _Float16 bl[C17_STEPS * 512];        // 17,408 B

    int n = blockIdx.z, g = blockIdx.y, yo0 = blockIdx.x * 8;
    const _Float16* plane = in_planes + ((size_t)(n * ncin + g) << 16);

    int iy0 = yo0 * 8 - 8;
    for (int i = threadIdx.x; i < C17_ROWS * 280; i += 256) {
        int r = i / 280, c = i % 280;
        int iy = iy0 + r, ix = c - 8;
        _Float16 v = (_Float16)0.f;
        if (iy >= 0 && iy < HW && ix >= 0 && ix < HW) v = plane[iy * HW + ix];
        tile[r * C17_COLS + c] = v;
    }
    for (int i = threadIdx.x; i < C17_STEPS * 256; i += 256)
        ((uint32_t*)bl)[i] = ((const uint32_t*)bf2)[i];
    __syncthreads();

    int lane = threadIdx.x & 31, wave = threadIdx.x >> 5;
    int mrow = lane & 15, off = (lane >> 4) * 8;

    for (int seg = wave; seg < 16; seg += 8) {
        int yo   = yo0 + (seg >> 1);
        int xo0  = (seg & 1) * 16;
        int xpix = xo0 + mrow;
        v8f c = {};
        #pragma unroll
        for (int s = 0; s < C17_STEPS; ++s) {
            int lr = (yo - yo0) * 8 + s;               // kernel row u == s
            const _Float16* p = &tile[lr * C17_COLS + 8 * xpix + off];
            v8h lo = *(const v8h*)p;                   // v = off + 0..7
            v8h hi = *(const v8h*)(p + 16);            // v = 16 + off + 0..7
            v16h a = __builtin_shufflevector(lo, hi, 0, 1, 2, 3, 4, 5, 6, 7,
                                             8, 9, 10, 11, 12, 13, 14, 15);
            v16h b = *(const v16h*)&bl[(s * 32 + lane) * 16];
            c = __builtin_amdgcn_wmma_f32_16x16x32_f16(false, a, false, b,
                                                       (short)0, c, false, false);
        }
        int j = lane & 15;
        if (j < NF) {
            int outch = ch_base + g * NF + j;
            float* obase = out + ((size_t)n * NCH + outch) * 1024;
            #pragma unroll
            for (int rr = 0; rr < 8; ++rr) {
                int m = rr + off;
                obase[yo * OUTW + xo0 + m] = c[rr];
            }
        }
    }
}

// ---------------- launcher ----------------
extern "C" void kernel_launch(void* const* d_in, const int* in_sizes, int n_in,
                              void* d_out, int out_size, void* d_ws, size_t ws_size,
                              hipStream_t stream)
{
    const float* img    = (const float*)d_in[0];   // [8,1,256,256]
    const float* psi_re = (const float*)d_in[1];   // [10,1,11,11]
    const float* psi_im = (const float*)d_in[2];   // [10,1,11,11]
    const float* blur_k = (const float*)d_in[3];   // [7,7]
    float* out = (float*)d_out;                    // [8,111,32,32] flat

    char* ws = (char*)d_ws;
    _Float16* img_h  = (_Float16*)(ws + IMG_H_OFF);
    _Float16* s1_abs = (_Float16*)(ws + S1ABS_OFF);
    _Float16* bf1re  = (_Float16*)(ws + BF1RE_OFF);
    _Float16* bf1im  = (_Float16*)(ws + BF1IM_OFF);
    _Float16* bf2    = (_Float16*)(ws + BF2_OFF);
    float*    C1     = (float*)(ws + C1_OFF);

    // 1) composed kernels + WMMA B fragments
    prep_kernel<<<1, 256, 0, stream>>>(psi_re, psi_im, blur_k, C1, bf1re, bf1im, bf2);

    // 2) image to f16
    int nelem = NB * HW * HW;
    img2h_kernel<<<(nelem + 255) / 256, 256, 0, stream>>>(img, img_h, nelem);

    // 3) s0 (channel 0)
    s0_kernel<<<(NB * 1024 + 255) / 256, 256, 0, stream>>>(img, blur_k, out);

    // 4) |s1| at full resolution (WMMA)
    s1_wmma_kernel<<<dim3(8, 8, NB), 256, 0, stream>>>(img_h, bf1re, bf1im, s1_abs);

    // 5) s1_out = (img (*) psi_re (*) blur)[::8]  -> channels 1..10
    conv17s8_wmma_kernel<<<dim3(4, 1, NB), 256, 0, stream>>>(img_h, bf2, out, 1, 1);

    // 6) s2_out = (|s1| (*) psi_re (*) blur)[::8] -> channels 11..110
    conv17s8_wmma_kernel<<<dim3(4, NF, NB), 256, 0, stream>>>(s1_abs, bf2, out, NF, 11);
}
